// MoE_16226386444690
// MI455X (gfx1250) — matrix operand order repaired and checked by reading, not compiled, not run on hardware
//
#include <hip/hip_runtime.h>
#include <hip/hip_bf16.h>

// MoE: T=8192 tokens, D=512, DI=256, E=8 experts, K=2 top-k.
// All GEMMs run on v_wmma_f32_16x16x32_bf16 (f32 data converted to bf16 while
// staging tiles into LDS; f32 accumulate). Expert weights stay L2-resident.

#define T_TOK 8192
#define D_IN  512
#define DI    256
#define NE    8
#define NPAIR (T_TOK * 2)        // 16384 (token, expert) pairs
#define BM    128                // M-tile (pairs per block)
#define KT    32                 // K-tile (matches WMMA K)
#define LDK   40                 // padded LDS row stride in bf16 elems (even)
#define MTILES (NPAIR / BM)      // 128 worst-case tiles per expert

typedef __attribute__((ext_vector_type(16))) __bf16 v16bf;
typedef __attribute__((ext_vector_type(8)))  float  v8f;

// ---- WMMA fragment gathers from row-major bf16 LDS tiles ------------------
// A-matrix 16x32 bf16 (ISA 7.12.2): lane m = L&15, half = L>>4,
// VGPR v packs k = (v>>2)*16 + half*8 + (v&3)*2 (+1).
__device__ __forceinline__ v16bf load_a_frag(const __bf16* row, int half) {
  union { v16bf v; unsigned u[8]; } f;
#pragma unroll
  for (int r = 0; r < 8; ++r) {
    const int k = ((r >> 2) << 4) + (half << 3) + ((r & 3) << 1);
    f.u[r] = *(const unsigned*)(row + k);   // even k -> dword aligned
  }
  return f.v;
}
// B-matrix 32x16 bf16: lane n = L&15, half = L>>4, VGPR v packs
// k = (v>>2)*16 + half*8 + (v&3)*2 (+1). `row` points at W[n][0] (row-major N x K weight),
// which is exactly B^T, i.e. B[k][n] = row[k].
__device__ __forceinline__ v16bf load_b_frag(const __bf16* row, int half) {
  union { v16bf v; unsigned u[8]; } f;
#pragma unroll
  for (int r = 0; r < 8; ++r) {
    const int k = ((r >> 2) << 4) + (half << 3) + ((r & 3) << 1);
    f.u[r] = *(const unsigned*)(row + k);
  }
  return f.v;
}

// ---------------------------------------------------------------------------
// Gate: one wave32 per token. scores = x @ wg^T -> softmax(8) -> top-2.
// meta[0..7] = per-expert pair counts (atomic).
__global__ __launch_bounds__(256)
void gate_kernel(const float* __restrict__ x, const float* __restrict__ wg,
                 int* __restrict__ meta, int* __restrict__ tokTop,
                 float* __restrict__ tokW) {
  __shared__ float wgl[NE * D_IN];                 // 16 KB
  for (int i = threadIdx.x; i < NE * D_IN; i += 256) wgl[i] = wg[i];
  __syncthreads();

  const int t    = blockIdx.x * 8 + (threadIdx.x >> 5);
  const int lane = threadIdx.x & 31;
  const float* xp = x + (size_t)t * D_IN;

  float acc[NE];
#pragma unroll
  for (int e = 0; e < NE; ++e) acc[e] = 0.0f;

#pragma unroll
  for (int j = 0; j < 4; ++j) {                    // 4 x (32 lanes x float4)
    const float4 xv = *(const float4*)(xp + j * 128 + lane * 4);
#pragma unroll
    for (int e = 0; e < NE; ++e) {
      const float* wp = wgl + e * D_IN + j * 128 + lane * 4;
      acc[e] = fmaf(xv.x, wp[0], fmaf(xv.y, wp[1],
               fmaf(xv.z, wp[2], fmaf(xv.w, wp[3], acc[e]))));
    }
  }
#pragma unroll
  for (int off = 16; off > 0; off >>= 1) {
#pragma unroll
    for (int e = 0; e < NE; ++e) acc[e] += __shfl_xor(acc[e], off, 32);
  }

  if (lane == 0) {
    float mx = acc[0];
#pragma unroll
    for (int e = 1; e < NE; ++e) mx = fmaxf(mx, acc[e]);
    float p[NE], sum = 0.0f;
#pragma unroll
    for (int e = 0; e < NE; ++e) { p[e] = __expf(acc[e] - mx); sum += p[e]; }
    const float inv = 1.0f / sum;
    int i0 = 0; float b0 = -1.0f;
#pragma unroll
    for (int e = 0; e < NE; ++e) { float v = p[e] * inv; if (v > b0) { b0 = v; i0 = e; } }
    int i1 = 0; float b1 = -1.0f;
#pragma unroll
    for (int e = 0; e < NE; ++e) {
      if (e == i0) continue;
      float v = p[e] * inv; if (v > b1) { b1 = v; i1 = e; }
    }
    tokTop[2 * t + 0] = i0;  tokW[2 * t + 0] = b0;
    tokTop[2 * t + 1] = i1;  tokW[2 * t + 1] = b1;
    atomicAdd(&meta[i0], 1);
    atomicAdd(&meta[i1], 1);
  }
}

// Exclusive scan of 8 counts -> meta[8..15]; zero cursors meta[16..23].
__global__ void scan_kernel(int* meta) {
  if (blockIdx.x == 0 && threadIdx.x == 0) {
    int o = 0;
#pragma unroll
    for (int e = 0; e < NE; ++e) { meta[8 + e] = o; o += meta[e]; meta[16 + e] = 0; }
  }
}

// Scatter pairs into globally compact per-expert slot lists.
__global__ __launch_bounds__(256)
void assign_kernel(const int* __restrict__ tokTop, const float* __restrict__ tokW,
                   int* __restrict__ meta, int* __restrict__ slotTok,
                   float* __restrict__ slotW) {
  const int t = blockIdx.x * 256 + threadIdx.x;
  if (t >= T_TOK) return;
#pragma unroll
  for (int k = 0; k < 2; ++k) {
    const int e   = tokTop[2 * t + k];
    const int pos = atomicAdd(&meta[16 + e], 1);
    const int s   = meta[8 + e] + pos;
    slotTok[s] = t;
    slotW[s]   = tokW[2 * t + k];
  }
}

// ---------------------------------------------------------------------------
// fc1 + SwiGLU: per block, 128 gathered token rows x full 512 cols of
// H = X @ fc1[e]^T, then a[:,j] = H[:,j] * silu(H[:,j+256]) stored bf16.
// 16 waves; wave w owns mfrag = w&7 (16 rows), nhalf = w>>3 (256 cols):
// 16 C-fragments (128 f32 acc VGPRs). Dynamic LDS: tiles aliased with H stage.
__global__ __launch_bounds__(512)
void fc1_kernel(const float* __restrict__ x, const float* __restrict__ fc1,
                const int* __restrict__ meta, const int* __restrict__ slotTok,
                __bf16* __restrict__ aBuf) {
  extern __shared__ unsigned char smem[];
  __bf16* Xl = (__bf16*)smem;                          // [BM][LDK]   10240 B
  __bf16* Bl = (__bf16*)(smem + BM * LDK * 2);         // [512][LDK]  40960 B
  __bf16* Hl = (__bf16*)smem;                          // [BM][512]  131072 B (alias)

  const int e     = blockIdx.y;
  const int cnt   = meta[e];
  const int tile0 = blockIdx.x * BM;
  if (tile0 >= cnt) return;
  const int base  = meta[8 + e];
  const int valid = (cnt - tile0 < BM) ? (cnt - tile0) : BM;

  const int tid   = threadIdx.x;
  const int wave  = tid >> 5;
  const int lane  = tid & 31;
  const int mfrag = wave & 7;
  const int nhalf = wave >> 3;
  const int lm    = lane & 15;
  const int lh    = lane >> 4;

  const float* fce = fc1 + (size_t)e * (2 * DI) * D_IN;

  v8f acc[16];
#pragma unroll
  for (int i = 0; i < 16; ++i)
#pragma unroll
    for (int j = 0; j < 8; ++j) acc[i][j] = 0.0f;

  // This thread's share of the X tile: row = tid>>2, 2 float4 cols.
  const int xrow = tid >> 2;
  const int c4   = (tid & 3) * 4;
  const int tok  = (xrow < valid) ? slotTok[base + tile0 + xrow] : -1;
  const float* xptr = (tok >= 0) ? (x + (size_t)tok * D_IN) : x;

  for (int kt = 0; kt < D_IN / KT; ++kt) {
    const int k0 = kt * KT;
    // X tile: 128x32 f32 -> bf16
    {
      __bf16* dst = Xl + xrow * LDK;
      if (tok >= 0) {
        const float4 v0 = *(const float4*)(xptr + k0 + c4);
        const float4 v1 = *(const float4*)(xptr + k0 + 16 + c4);
        dst[c4 + 0] = (__bf16)v0.x; dst[c4 + 1] = (__bf16)v0.y;
        dst[c4 + 2] = (__bf16)v0.z; dst[c4 + 3] = (__bf16)v0.w;
        dst[c4 + 16] = (__bf16)v1.x; dst[c4 + 17] = (__bf16)v1.y;
        dst[c4 + 18] = (__bf16)v1.z; dst[c4 + 19] = (__bf16)v1.w;
      } else {
        const __bf16 z = (__bf16)0.0f;
#pragma unroll
        for (int j = 0; j < 4; ++j) { dst[c4 + j] = z; dst[c4 + 16 + j] = z; }
      }
    }
    // fc1 tile: 512x32 f32 -> bf16 (8 float4 per thread)
#pragma unroll
    for (int i = 0; i < 8; ++i) {
      const int q   = tid + (i << 9);
      const int row = q >> 3;
      const int cc  = (q & 7) << 2;
      const float4 v = *(const float4*)(fce + (size_t)row * D_IN + k0 + cc);
      __bf16* dst = Bl + row * LDK + cc;
      dst[0] = (__bf16)v.x; dst[1] = (__bf16)v.y;
      dst[2] = (__bf16)v.z; dst[3] = (__bf16)v.w;
    }
    __syncthreads();

    const v16bf af = load_a_frag(Xl + (mfrag * 16 + lm) * LDK, lh);
#pragma unroll
    for (int nf = 0; nf < 16; ++nf) {
      const v16bf bf = load_b_frag(Bl + (nhalf * 256 + nf * 16 + lm) * LDK, lh);
      acc[nf] = __builtin_amdgcn_wmma_f32_16x16x32_bf16(
          false, af, false, bf, (short)0, acc[nf], false, false);
    }
    __syncthreads();
  }

  // Stage H (bf16) to LDS, aliasing the tile buffers (safe: synced above).
#pragma unroll
  for (int nf = 0; nf < 16; ++nf) {
    const int ncol = nhalf * 256 + nf * 16 + lm;
#pragma unroll
    for (int r = 0; r < 8; ++r) {
      const int m = mfrag * 16 + r + (lh << 3);
      Hl[m * 512 + ncol] = (__bf16)acc[nf][r];
    }
  }
  __syncthreads();

  // a = y * silu(g): 128x256 elems / 512 threads = 64 each.
#pragma unroll
  for (int i = 0; i < 64; ++i) {
    const int idx = tid + (i << 9);
    const int m = idx >> 8;
    const int j = idx & 255;
    if (m < valid) {
      const float y = (float)Hl[m * 512 + j];
      const float g = (float)Hl[m * 512 + 256 + j];
      const float s = g / (1.0f + __expf(-g));     // silu(g)
      aBuf[(size_t)(base + tile0 + m) * DI + j] = (__bf16)(y * s);
    }
  }
}

// ---------------------------------------------------------------------------
// fc2 + weighted combine: O = a @ fc2[e]^T (K=256), then
// out[token, d] += weight * O  via global f32 atomics (2 adds per element).
__global__ __launch_bounds__(512)
void fc2_kernel(const __bf16* __restrict__ aBuf, const float* __restrict__ fc2,
                const int* __restrict__ meta, const int* __restrict__ slotTok,
                const float* __restrict__ slotW, float* __restrict__ out) {
  __shared__ __bf16 Al[BM * LDK];      // 10240 B
  __shared__ __bf16 Bl[512 * LDK];     // 40960 B

  const int e     = blockIdx.y;
  const int cnt   = meta[e];
  const int tile0 = blockIdx.x * BM;
  if (tile0 >= cnt) return;
  const int base  = meta[8 + e];
  const int valid = (cnt - tile0 < BM) ? (cnt - tile0) : BM;

  const int tid   = threadIdx.x;
  const int wave  = tid >> 5;
  const int lane  = tid & 31;
  const int mfrag = wave & 7;
  const int nhalf = wave >> 3;
  const int lm    = lane & 15;
  const int lh    = lane >> 4;

  const float* fce = fc2 + (size_t)e * D_IN * DI;

  v8f acc[16];
#pragma unroll
  for (int i = 0; i < 16; ++i)
#pragma unroll
    for (int j = 0; j < 8; ++j) acc[i][j] = 0.0f;

  const int arow = tid >> 2;           // a-tile: 128 rows x 32 bf16, 8 per thread
  const int c8   = (tid & 3) * 8;

  for (int kt = 0; kt < DI / KT; ++kt) {
    const int k0 = kt * KT;
    // a tile (already bf16): 16B per thread
    {
      __bf16* dst = Al + arow * LDK + c8;
      if (arow < valid) {
        const __bf16* src = aBuf + (size_t)(base + tile0 + arow) * DI + k0 + c8;
#pragma unroll
        for (int j = 0; j < 8; ++j) dst[j] = src[j];
      } else {
        const __bf16 z = (__bf16)0.0f;
#pragma unroll
        for (int j = 0; j < 8; ++j) dst[j] = z;
      }
    }
    // fc2 tile: 512x32 f32 -> bf16
#pragma unroll
    for (int i = 0; i < 8; ++i) {
      const int q   = tid + (i << 9);
      const int row = q >> 3;
      const int cc  = (q & 7) << 2;
      const float4 v = *(const float4*)(fce + (size_t)row * DI + k0 + cc);
      __bf16* dst = Bl + row * LDK + cc;
      dst[0] = (__bf16)v.x; dst[1] = (__bf16)v.y;
      dst[2] = (__bf16)v.z; dst[3] = (__bf16)v.w;
    }
    __syncthreads();

    const v16bf af = load_a_frag(Al + (mfrag * 16 + lm) * LDK, lh);
#pragma unroll
    for (int nf = 0; nf < 16; ++nf) {
      const v16bf bf = load_b_frag(Bl + (nhalf * 256 + nf * 16 + lm) * LDK, lh);
      acc[nf] = __builtin_amdgcn_wmma_f32_16x16x32_bf16(
          false, af, false, bf, (short)0, acc[nf], false, false);
    }
    __syncthreads();
  }

  // Scatter: cache this lane's 8 row descriptors, then weighted atomics.
  int   toks[8];
  float wts[8];
  bool  vr[8];
#pragma unroll
  for (int r = 0; r < 8; ++r) {
    const int m = mfrag * 16 + r + (lh << 3);
    vr[r] = (m < valid);
    const int s = base + tile0 + (vr[r] ? m : 0);
    toks[r] = vr[r] ? slotTok[s] : 0;
    wts[r]  = vr[r] ? slotW[s]  : 0.0f;
  }
#pragma unroll
  for (int nf = 0; nf < 16; ++nf) {
    const int d = nhalf * 256 + nf * 16 + lm;
#pragma unroll
    for (int r = 0; r < 8; ++r) {
      if (vr[r])
        atomicAdd(out + (size_t)toks[r] * D_IN + d, wts[r] * acc[nf][r]);
    }
  }
}

// ---------------------------------------------------------------------------
extern "C" void kernel_launch(void* const* d_in, const int* in_sizes, int n_in,
                              void* d_out, int out_size, void* d_ws, size_t ws_size,
                              hipStream_t stream) {
  const float* x   = (const float*)d_in[0];
  const float* wg  = (const float*)d_in[1];
  const float* fc1 = (const float*)d_in[2];
  const float* fc2 = (const float*)d_in[3];
  float* out = (float*)d_out;

  // Workspace carve (~8.65 MB total):
  unsigned char* ws = (unsigned char*)d_ws;
  int*    meta    = (int*)ws;                                   // 256 B (counts/offsets/cursors)
  int*    slotTok = (int*)(ws + 256);                           // 64 KB
  float*  slotW   = (float*)(ws + 256 + 1 * 65536);             // 64 KB
  int*    tokTop  = (int*)(ws + 256 + 2 * 65536);               // 64 KB
  float*  tokW    = (float*)(ws + 256 + 3 * 65536);             // 64 KB
  __bf16* aBuf    = (__bf16*)(ws + 256 + 4 * 65536);            // 8 MB (16384 x 256 bf16)

  hipMemsetAsync(out, 0, (size_t)out_size * sizeof(float), stream);
  hipMemsetAsync(meta, 0, 256, stream);

  gate_kernel<<<T_TOK / 8, 256, 0, stream>>>(x, wg, meta, tokTop, tokW);
  scan_kernel<<<1, 32, 0, stream>>>(meta);
  assign_kernel<<<T_TOK / 256, 256, 0, stream>>>(tokTop, tokW, meta, slotTok, slotW);

  fc1_kernel<<<dim3(MTILES, NE), 512, 131072, stream>>>(x, fc1, meta, slotTok, aBuf);
  fc2_kernel<<<dim3(MTILES, NE), 512, 0, stream>>>(aBuf, fc2, meta, slotTok, slotW, out);
}
